// W4A16Linear_20392504721854
// MI455X (gfx1250) — matrix operand order repaired and checked by reading, not compiled, not run on hardware
//
#include <hip/hip_runtime.h>

// ---------------- problem constants ----------------
#define M_DIM 256
#define K_DIM 4096
#define N_DIM 11008
#define GSZ   128

// ---------------- tiling ----------------
#define BM 256   // full token dim in one block column: each int4 weight dequantized once
#define BN 64
#define BK 64
#define NKB (K_DIM / BK)   // 64 K-iterations
#define LDA 72   // padded LDS pitch (elements) for A rows   (BK + 8) -> 144B, 16B-aligned rows
#define LDW 72   // padded LDS pitch (elements) for Wt rows  (BK + 8)

typedef __attribute__((ext_vector_type(16))) __bf16        v16bf;
typedef __attribute__((ext_vector_type(2)))  __bf16        v2bf;
typedef __attribute__((ext_vector_type(8)))  float         v8f;
typedef __attribute__((ext_vector_type(8)))  unsigned int  u32x8;
typedef __attribute__((ext_vector_type(4)))  unsigned int  u32x4;
typedef __attribute__((ext_vector_type(2)))  unsigned int  u32x2;

__device__ __forceinline__ float bf16bits_to_f(unsigned short b) {
    unsigned int u = ((unsigned int)b) << 16;
    return __builtin_bit_cast(float, u);
}

// packed f32x2 -> bf16x2 (RNE fptrunc pair -> v_cvt_pk_bf16_f32 on gfx1250)
__device__ __forceinline__ unsigned int pack2_bf16(float a, float b) {
    v2bf v;
    v[0] = (__bf16)a;
    v[1] = (__bf16)b;
    return __builtin_bit_cast(unsigned int, v);
}

__device__ __forceinline__ v16bf make_frag(u32x4 a, u32x4 b) {
    u32x8 t;
    t[0] = a[0]; t[1] = a[1]; t[2] = a[2]; t[3] = a[3];
    t[4] = b[0]; t[5] = b[1]; t[6] = b[2]; t[7] = b[3];
    return __builtin_bit_cast(v16bf, t);
}

// per-tile staging registers (loads issued one iteration ahead)
struct TileRegs {
    u32x4 a[8];      // A tile chunks: 8 x 16B per thread
    u32x4 q0, q1;    // packed int4 weights, 2 passes of 4 columns
    u32x2 sc, zr;    // bf16 scales / zeros for this thread's 4 columns
};

__device__ __forceinline__ void issue_tile_loads(
    TileRegs& t, int kb,
    const unsigned short* __restrict__ x,
    const int*            __restrict__ Wq,
    const unsigned short* __restrict__ scales,
    const unsigned short* __restrict__ zeros,
    int n0, int a_chunk, int a_row0, int w_c0, int w_r0)
{
    const int k0 = kb * BK;
    const int g  = k0 / GSZ;
    #pragma unroll
    for (int p = 0; p < 8; ++p) {
        const int row = a_row0 + 32 * p;
        t.a[p] = *(const u32x4*)(x + (size_t)row * K_DIM + k0 + a_chunk * 8);
    }
    t.sc = *(const u32x2*)(scales + (size_t)g * N_DIM + n0 + w_c0);
    t.zr = *(const u32x2*)(zeros  + (size_t)g * N_DIM + n0 + w_c0);
    t.q0 = *(const u32x4*)(Wq + (size_t)(k0 / 2 + w_r0)      * N_DIM + n0 + w_c0);
    t.q1 = *(const u32x4*)(Wq + (size_t)(k0 / 2 + w_r0 + 16) * N_DIM + n0 + w_c0);
}

__global__ __launch_bounds__(256)
void W4A16Linear_gfx1250_kernel(const unsigned short* __restrict__ x,      // (M,K) bf16
                                const int*            __restrict__ Wq,     // (K/2,N) int32, 2 nibbles
                                const unsigned short* __restrict__ scales, // (NG,N) bf16
                                const unsigned short* __restrict__ zeros,  // (NG,N) bf16
                                const unsigned short* __restrict__ bias,   // (N,) bf16
                                unsigned short*       __restrict__ out)    // (M,N) bf16
{
    __shared__ unsigned short As[BM * LDA];   // A tile, row-major (m, k)     : 36 KB
    __shared__ unsigned short Ws[BN * LDW];   // dequantized W tile, (n, k)   :  9 KB

    const int tid  = threadIdx.x;
    const int lane = tid & 31;
    const int wave = tid >> 5;     // 0..7 : wave owns rows [wave*32, wave*32+32), all 64 cols
    const int half = lane >> 4;    // 0..1
    const int lr   = lane & 15;    // 0..15

    const int n0 = blockIdx.x * BN;

    // A staging: 8 passes, each thread loads one 16B chunk of a row
    const int a_chunk = tid & 7;   // 8 x 16B chunks cover BK=64 bf16
    const int a_row0  = tid >> 3;  // 0..31, +32 per pass

    // W staging: each thread handles 4 consecutive N columns, 2 packed-K rows
    const int w_c0 = (tid & 15) * 4;  // 0..60
    const int w_r0 = tid >> 4;        // 0..15, +16 per pass

    v8f acc[2][4] = {};   // 2 M-tiles x 4 N-tiles of 16x16 f32

    TileRegs t;
    issue_tile_loads(t, 0, x, Wq, scales, zeros, n0, a_chunk, a_row0, w_c0, w_r0);

    for (int kb = 0; kb < NKB; ++kb) {
        // ---- dequantize this tile's weights from registers (waits on in-flight loads) ----
        float s[4], zs[4];
        #pragma unroll
        for (int j = 0; j < 4; ++j) {
            const float sv = bf16bits_to_f((unsigned short)(t.sc[j >> 1] >> (16 * (j & 1))));
            const float zv = bf16bits_to_f((unsigned short)(t.zr[j >> 1] >> (16 * (j & 1))));
            s[j]  = sv;
            zs[j] = zv * sv;   // (nib + z)*s == fma(nib, s, z*s)
        }
        unsigned int wpk[2][4];
        #pragma unroll
        for (int p = 0; p < 2; ++p) {
            const u32x4 q = p ? t.q1 : t.q0;
            #pragma unroll
            for (int j = 0; j < 4; ++j) {
                const unsigned int w = q[j];
                const float lo = (float)(w & 15u);         // K = 2r   (low nibble first)
                const float hi = (float)((w >> 4) & 15u);  // K = 2r+1
                wpk[p][j] = pack2_bf16(__builtin_fmaf(lo, s[j], zs[j]),
                                       __builtin_fmaf(hi, s[j], zs[j]));
            }
        }

        __syncthreads();   // previous compute phase done reading LDS

        // ---- store staged tile into LDS ----
        #pragma unroll
        for (int p = 0; p < 8; ++p) {
            const int row = a_row0 + 32 * p;
            *(u32x4*)(&As[row * LDA + a_chunk * 8]) = t.a[p];
        }
        #pragma unroll
        for (int p = 0; p < 2; ++p) {
            const int r = w_r0 + 16 * p;   // packed-K row within tile (0..31)
            #pragma unroll
            for (int j = 0; j < 4; ++j) {
                *(unsigned int*)(&Ws[(w_c0 + j) * LDW + 2 * r]) = wpk[p][j];
            }
        }

        __syncthreads();   // staging visible to all waves

        // ---- issue next tile's global loads: latency hidden under WMMA compute ----
        if (kb + 1 < NKB) {
            issue_tile_loads(t, kb + 1, x, Wq, scales, zeros,
                             n0, a_chunk, a_row0, w_c0, w_r0);
        }
        // warm L2 two tiles ahead (global_prefetch_b8)
        if (kb + 2 < NKB) {
            __builtin_prefetch(Wq + (size_t)((kb + 2) * (BK / 2) + w_r0) * N_DIM + n0 + w_c0, 0, 1);
        }

        // ---------------- compute: 2 WMMA k-steps of 32 ----------------
        #pragma unroll
        for (int ks = 0; ks < 2; ++ks) {
            const int kk = ks * 32;
            v16bf afrag[2], bfrag[4];

            #pragma unroll
            for (int tm = 0; tm < 2; ++tm) {
                // A 16x32 bf16 layout: lanes 0-15: K 0..7 & 16..23; lanes 16-31: K 8..15 & 24..31
                const int row = wave * 32 + tm * 16 + lr;
                u32x4 c0 = *(const u32x4*)(&As[row * LDA + kk + half * 8]);
                u32x4 c1 = *(const u32x4*)(&As[row * LDA + kk + 16 + half * 8]);
                afrag[tm] = make_frag(c0, c1);
            }
            #pragma unroll
            for (int tn = 0; tn < 4; ++tn) {
                // B 32x16 bf16 layout: lane = column; lanes 0-15: K 0..7 & 16..23; lanes 16-31: K 8..15 & 24..31
                const int col = tn * 16 + lr;
                u32x4 b0 = *(const u32x4*)(&Ws[col * LDW + kk + half * 8]);
                u32x4 b1 = *(const u32x4*)(&Ws[col * LDW + kk + 16 + half * 8]);
                bfrag[tn] = make_frag(b0, b1);
            }

            #pragma unroll
            for (int tm = 0; tm < 2; ++tm) {
                #pragma unroll
                for (int tn = 0; tn < 4; ++tn) {
                    acc[tm][tn] = __builtin_amdgcn_wmma_f32_16x16x32_bf16(
                        false, afrag[tm], false, bfrag[tn],
                        (short)0, acc[tm][tn], false, false);
                }
            }
        }
    }

    // ---------------- epilogue: + bias, packed f32->bf16, store ----------------
    #pragma unroll
    for (int tn = 0; tn < 4; ++tn) {
        const int col = n0 + tn * 16 + lr;
        const float bf = bf16bits_to_f(bias[col]);
        #pragma unroll
        for (int tm = 0; tm < 2; ++tm) {
            const int rbase = wave * 32 + tm * 16 + half * 8;  // VGPR v -> row 8*half + v
            #pragma unroll
            for (int v = 0; v < 8; v += 2) {
                const unsigned int u = pack2_bf16(acc[tm][tn][v]     + bf,
                                                  acc[tm][tn][v + 1] + bf);
                out[(size_t)(rbase + v)     * N_DIM + col] = (unsigned short)(u & 0xFFFFu);
                out[(size_t)(rbase + v + 1) * N_DIM + col] = (unsigned short)(u >> 16);
            }
        }
    }
}

extern "C" void kernel_launch(void* const* d_in, const int* in_sizes, int n_in,
                              void* d_out, int out_size, void* d_ws, size_t ws_size,
                              hipStream_t stream) {
    (void)in_sizes; (void)n_in; (void)out_size; (void)d_ws; (void)ws_size;
    const unsigned short* x  = (const unsigned short*)d_in[0];
    const int*            Wq = (const int*)d_in[1];
    const unsigned short* sc = (const unsigned short*)d_in[2];
    const unsigned short* zr = (const unsigned short*)d_in[3];
    const unsigned short* bi = (const unsigned short*)d_in[4];
    unsigned short*       y  = (unsigned short*)d_out;

    dim3 grid(N_DIM / BN, 1);   // 172 workgroups, each covers all 256 token rows
    dim3 block(256);            // 8 wave32
    W4A16Linear_gfx1250_kernel<<<grid, block, 0, stream>>>(x, Wq, sc, zr, bi, y);
}
